// TGAT_89558658056628
// MI455X (gfx1250) — matrix operand order repaired and checked by reading, not compiled, not run on hardware
//
#include <hip/hip_runtime.h>
#include <math.h>

// ---------------- problem constants ----------------
#define NN      10000   // nodes
#define KNBR    32      // neighbors per node
#define HF      128     // hidden
#define HEADS   8
#define HD      16
#define TF      64      // time feature
#define RF      32      // rel feature
#define OUTF    64
#define KQV3    384
#define START_T 0.25f
#define END_T   0.75f
#define ATTN_NORM 0.25f              // sqrt(1/16)
#define T_NORM  0.17677669529663689f // sqrt(1/32)

typedef __attribute__((ext_vector_type(16))) _Float16 v16h;
typedef __attribute__((ext_vector_type(8)))  float    v8f;

union FragU { v16h h; unsigned u[8]; };

// ---------------- weight packing into WMMA B-fragment order ----------------
// dst layout: tile (kc, nt) -> 32 lanes x 16 halves contiguous:
//   dst[(((kc*NT)+nt)*32 + lane)*16 + e]
// where lane half = lane>>4, l16 = lane&15, and element e maps to
//   k = kc*32 + (e<8 ? (lane>>4)*8 + e : 16 + (lane>>4)*8 + (e-8)),  n = nt*16 + l16
// colskip!=0 implements the k/v column split (skip the q chunk of a [*,384] weight).
__global__ void pack_frag_kernel(const float* __restrict__ src, int ldsrc, int colskip,
                                 int NT, int total, _Float16* __restrict__ dst) {
  int idx = blockIdx.x * 256 + threadIdx.x;
  if (idx >= total) return;
  int e    = idx & 15;
  int lane = (idx >> 4) & 31;
  int rest = idx >> 9;
  int nt   = rest % NT;
  int kc   = rest / NT;
  int k = kc * 32 + ((e < 8) ? ((lane >> 4) * 8 + e) : (16 + (lane >> 4) * 8 + (e - 8)));
  int n = nt * 16 + (lane & 15);
  int sc = (n < 128) ? n : (n + colskip);
  dst[idx] = (_Float16)src[(size_t)k * ldsrc + sc];
}

// ---------------- generic 1-wave WMMA GEMM ----------------
// C[M,N] = epi(A_f32[M,K] @ Bfrag_f16[K,N]); grid=(M/16, N/16), block=32 (one wave)
template <bool HAS_BIAS, bool DO_RELU, bool HAS_RES>
__global__ __launch_bounds__(32)
void wmma_gemm_kernel(const float* __restrict__ A, int lda,
                      const _Float16* __restrict__ B, int NT,
                      const float* __restrict__ bias,
                      const float* __restrict__ resid, int ldr,
                      float* __restrict__ C, int ldc,
                      int Kdim)
{
  const int lane = threadIdx.x;
  const int half = lane >> 4;
  const int l16  = lane & 15;
  const int rowA = blockIdx.x * 16 + l16;
  const int colB = blockIdx.y * 16 + l16;

  const float* Arow = A + (size_t)rowA * lda;
  v8f acc = {};
  for (int k0 = 0; k0 < Kdim; k0 += 32) {
    if (k0 + 32 < Kdim) __builtin_prefetch(Arow + k0 + 32, 0, 1);
    v16h a;
#pragma unroll
    for (int v = 0; v < 8; ++v) {
      int kb = (v < 4) ? (half * 8 + 2 * v) : (16 + half * 8 + 2 * (v - 4));
      float2 p = *(const float2*)(Arow + k0 + kb);
      a[2 * v]     = (_Float16)p.x;
      a[2 * v + 1] = (_Float16)p.y;
    }
    // fragment-ordered B: one contiguous 32B chunk per lane
    v16h b = *(const v16h*)(B + ((size_t)(((k0 >> 5) * NT) + blockIdx.y) * 32 + lane) * 16);
    acc = __builtin_amdgcn_wmma_f32_16x16x32_f16(false, a, false, b, (short)0, acc, false, false);
  }

  const float bcol = HAS_BIAS ? bias[colB] : 0.f;
  const int r0 = blockIdx.x * 16 + half * 8;
  float* cp = C + (size_t)r0 * ldc + colB;
  const float* rp = HAS_RES ? (resid + (size_t)r0 * ldr + colB) : nullptr;
#pragma unroll
  for (int i = 0; i < 8; ++i) {
    float v = acc[i] + bcol;
    if (DO_RELU) v = fmaxf(v, 0.f);
    if (HAS_RES) v += rp[(size_t)i * ldr];
    cp[(size_t)i * ldc] = v;
  }
}

// ---------------- layernorm (block per row of 128) ----------------
__global__ __launch_bounds__(128)
void ln_kernel(const float* __restrict__ x, const float* __restrict__ g,
               const float* __restrict__ bb, float* __restrict__ out1, int ld1)
{
  __shared__ float sred[128];
  int row = blockIdx.x, tid = threadIdx.x;
  float v = x[(size_t)row * HF + tid];
  sred[tid] = v; __syncthreads();
  for (int s = 64; s > 0; s >>= 1) { if (tid < s) sred[tid] += sred[tid + s]; __syncthreads(); }
  float mean = sred[0] * (1.0f / HF);
  __syncthreads();
  float d = v - mean;
  sred[tid] = d * d; __syncthreads();
  for (int s = 64; s > 0; s >>= 1) { if (tid < s) sred[tid] += sred[tid + s]; __syncthreads(); }
  float var = sred[0] * (1.0f / HF);
  float y = d * rsqrtf(var + 1e-5f) * g[tid] + bb[tid];
  out1[(size_t)row * ld1 + tid] = y;
}

// ---------------- time-query: q = q_ + (t2v(tmax-start) @ Wtime)[:,H:2H] ----------------
__global__ __launch_bounds__(128)
void timeq_kernel(const float* __restrict__ times, const float* __restrict__ Wt,
                  const float* __restrict__ bt, const float* __restrict__ Wtime,
                  const float* __restrict__ kqv, float* __restrict__ qbuf)
{
  __shared__ float tq[TF];
  __shared__ float tmaxS;
  int n = blockIdx.x, tid = threadIdx.x;
  if (tid < 32) {
    float t = times[n * KNBR + tid];
    int m = (t >= START_T) && (t < END_T);
    float v = m ? t : START_T;
    for (int off = 16; off > 0; off >>= 1) v = fmaxf(v, __shfl_xor(v, off, 32));
    if (tid == 0) tmaxS = v;
  }
  __syncthreads();
  if (tid < 32) {
    float hv = (tmaxS - START_T) * Wt[tid] + bt[tid];
    tq[2 * tid]     = __sinf(hv) * T_NORM;
    tq[2 * tid + 1] = __cosf(hv) * T_NORM;
  }
  __syncthreads();
  float s = 0.f;
#pragma unroll 8
  for (int f = 0; f < TF; ++f) s += tq[f] * Wtime[f * KQV3 + HF + tid];
  qbuf[(size_t)n * HF + tid] = kqv[(size_t)n * KQV3 + HF + tid] + s;
}

// ---------------- fused neighbor-feature + attention kernel (block per node) ----------------
__global__ __launch_bounds__(256)
void attn_kernel(const float* __restrict__ times, const int* __restrict__ nbrs,
                 const float* __restrict__ rels, const float* __restrict__ Wt,
                 const float* __restrict__ bt,
                 const _Float16* __restrict__ Wtkv, const _Float16* __restrict__ Wekv,
                 const float* __restrict__ kqv, const float* __restrict__ qbuf,
                 const float* __restrict__ xn, int xnld, float* __restrict__ h2)
{
  __shared__ _Float16 At[KNBR][TF];   // tn_emb (f16)       4 KB
  __shared__ _Float16 Ar[KNBR][RF];   // rels   (f16)       2 KB
  __shared__ float kvS[2][KNBR][HF];  // k/v accumulators  32 KB
  __shared__ float qS[HF];
  __shared__ float timesS[KNBR];
  __shared__ int   maskS[KNBR];
  __shared__ float tmaxS;
  __shared__ int   hasS;

  int n = blockIdx.x, tid = threadIdx.x;
  if (tid < 32) {
    float t = times[n * KNBR + tid];
    int m = (t >= START_T) && (t < END_T);
    timesS[tid] = t; maskS[tid] = m;
    float v = m ? t : START_T;
    int hv = m;
    for (int off = 16; off > 0; off >>= 1) {
      v = fmaxf(v, __shfl_xor(v, off, 32));
      hv |= __shfl_xor(hv, off, 32);
    }
    if (tid == 0) { tmaxS = v; hasS = hv; }
  }
  if (tid < HF) qS[tid] = qbuf[(size_t)n * HF + tid];
  __syncthreads();

  float tmax = tmaxS;
  // time embeddings -> f16 LDS (WMMA A operand, row-major)
  for (int idx = tid; idx < KNBR * 32; idx += 256) {
    int j = idx >> 5, m = idx & 31;
    float dt = tmax - timesS[j];
    float hv = dt * Wt[m] + bt[m];
    At[j][2 * m]     = (_Float16)(__sinf(hv) * T_NORM);
    At[j][2 * m + 1] = (_Float16)(__cosf(hv) * T_NORM);
  }
  // rels -> f16 LDS
  for (int idx = tid; idx < KNBR * RF; idx += 256) {
    int j = idx >> 5, c = idx & 31;
    Ar[j][c] = (_Float16)rels[((size_t)n * KNBR + j) * RF + c];
  }
  // gather kn / vn into LDS as C-matrix init
  for (int idx = tid; idx < KNBR * HF; idx += 256) {
    int j = idx >> 7, c = idx & 127;
    int nb = nbrs[n * KNBR + j];
    kvS[0][j][c] = kqv[(size_t)nb * KQV3 + c];
    kvS[1][j][c] = kqv[(size_t)nb * KQV3 + 256 + c];
  }
  __syncthreads();

  // WMMA: kv(32x256) += At(32x64)@Wtkv(64x256) + Ar(32x32)@Wekv(32x256)
  int wave = tid >> 5, lane = tid & 31;
  int half = lane >> 4, l16 = lane & 15;
  for (int t = wave; t < 32; t += 8) {          // wave-uniform loop; EXEC stays full
    int mt  = t >> 4;
    int nt  = t & 15;
    int buf = nt >> 3;
    int cb  = (nt & 7) * 16;
    v8f acc;
#pragma unroll
    for (int i = 0; i < 8; ++i)
      acc[i] = kvS[buf][mt * 16 + half * 8 + i][cb + l16];
    int rowA = mt * 16 + l16;
#pragma unroll
    for (int kc = 0; kc < 2; ++kc) {
      int k0 = kc * 32;
      FragU ua;
#pragma unroll
      for (int v = 0; v < 8; ++v) {
        int kb = (v < 4) ? (half * 8 + 2 * v) : (16 + half * 8 + 2 * (v - 4));
        ua.u[v] = *(const unsigned*)&At[rowA][k0 + kb];  // pair (k, k+1)
      }
      v16h b = *(const v16h*)(Wtkv + ((size_t)((kc * 16) + nt) * 32 + lane) * 16);
      acc = __builtin_amdgcn_wmma_f32_16x16x32_f16(false, ua.h, false, b, (short)0, acc, false, false);
    }
    {
      FragU ua;
#pragma unroll
      for (int v = 0; v < 8; ++v) {
        int kb = (v < 4) ? (half * 8 + 2 * v) : (16 + half * 8 + 2 * (v - 4));
        ua.u[v] = *(const unsigned*)&Ar[rowA][kb];
      }
      v16h b = *(const v16h*)(Wekv + ((size_t)nt * 32 + lane) * 16);
      acc = __builtin_amdgcn_wmma_f32_16x16x32_f16(false, ua.h, false, b, (short)0, acc, false, false);
    }
#pragma unroll
    for (int i = 0; i < 8; ++i)
      kvS[buf][mt * 16 + half * 8 + i][cb + l16] = acc[i];
  }
  __syncthreads();

  // attention: wave == head, lane == neighbor
  int hId = wave;
  int j = lane;
  float score = 0.f;
#pragma unroll
  for (int d = 0; d < HD; ++d) score += qS[hId * HD + d] * kvS[0][j][hId * HD + d];
  score *= ATTN_NORM;
  if (!maskS[j]) score = -1e30f;
  float mx = score;
  for (int off = 16; off > 0; off >>= 1) mx = fmaxf(mx, __shfl_xor(mx, off, 32));
  float ex = __expf(score - mx);
  float sm = ex;
  for (int off = 16; off > 0; off >>= 1) sm += __shfl_xor(sm, off, 32);
  float attn = ex / sm;

  int d = lane & 15;
  float od = 0.f;
  for (int jj = 0; jj < KNBR; ++jj) {
    float aj = __shfl(attn, jj, 32);
    od += aj * kvS[1][jj][hId * HD + d];
  }
  if (lane < HD) {
    int c = hId * HD + d;
    float o = hasS ? od : 0.f;
    h2[(size_t)n * HF + c] = o + xn[(size_t)n * xnld + c];
  }
}

// ---------------- host orchestration ----------------
extern "C" void kernel_launch(void* const* d_in, const int* in_sizes, int n_in,
                              void* d_out, int out_size, void* d_ws, size_t ws_size,
                              hipStream_t stream) {
  (void)in_sizes; (void)n_in; (void)out_size; (void)ws_size;

  const float* x      = (const float*)d_in[0];
  const int*   nbrs   = (const int*)  d_in[1];
  const float* times  = (const float*)d_in[2];
  const float* rels   = (const float*)d_in[3];
  const float* Wp     = (const float*)d_in[4];
  const float* bp     = (const float*)d_in[5];
  const float* Wkqv   = (const float*)d_in[6];
  const float* Wt     = (const float*)d_in[7];
  const float* bt     = (const float*)d_in[8];
  const float* Wtime  = (const float*)d_in[9];
  const float* Wedge  = (const float*)d_in[10];
  const float* g1w    = (const float*)d_in[11];
  const float* b1n    = (const float*)d_in[12];
  const float* g2w    = (const float*)d_in[13];
  const float* b2n    = (const float*)d_in[14];
  const float* W1     = (const float*)d_in[15];
  const float* bl1    = (const float*)d_in[16];
  const float* W2     = (const float*)d_in[17];
  const float* bl2    = (const float*)d_in[18];
  const float* Wo     = (const float*)d_in[19];
  const float* bo     = (const float*)d_in[20];
  float* out = (float*)d_out;

  // workspace layout (256B aligned)
  char* ws = (char*)d_ws;
  size_t off = 0;
  auto alloc = [&](size_t bytes) { size_t o = off; off = (off + bytes + 255) & ~(size_t)255; return o; };
  float* hA      = (float*)(ws + alloc((size_t)NN * HF * 4));
  float* hB      = (float*)(ws + alloc((size_t)NN * HF * 4));
  float* kqvB    = (float*)(ws + alloc((size_t)NN * KQV3 * 4));
  float* qbuf    = (float*)(ws + alloc((size_t)NN * HF * 4));
  float* h2B     = (float*)(ws + alloc((size_t)NN * HF * 4));
  float* concatB = (float*)(ws + alloc((size_t)NN * 256 * 4));   // [xn | hn]
  float* z1      = (float*)(ws + alloc((size_t)NN * HF * 4));
  _Float16* Wp16   = (_Float16*)(ws + alloc(128 * 128 * 2));
  _Float16* Wkqv16 = (_Float16*)(ws + alloc(128 * 384 * 2));
  _Float16* W116   = (_Float16*)(ws + alloc(256 * 128 * 2));
  _Float16* W216   = (_Float16*)(ws + alloc(128 * 128 * 2));
  _Float16* Wo16   = (_Float16*)(ws + alloc(128 * 64 * 2));
  _Float16* Wtkv16 = (_Float16*)(ws + alloc(64 * 256 * 2));
  _Float16* Wekv16 = (_Float16*)(ws + alloc(32 * 256 * 2));

  auto pack = [&](const float* s, int ldsrc, int colskip, int NT, int total, _Float16* d) {
    pack_frag_kernel<<<(total + 255) / 256, 256, 0, stream>>>(s, ldsrc, colskip, NT, total, d);
  };
  pack(Wp,    128, 0,   8,  128 * 128, Wp16);
  pack(Wkqv,  384, 0,   24, 128 * 384, Wkqv16);
  pack(W1,    128, 0,   8,  256 * 128, W116);
  pack(W2,    128, 0,   8,  128 * 128, W216);
  pack(Wo,    64,  0,   4,  128 * 64,  Wo16);
  pack(Wtime, 384, 128, 16, 64 * 256,  Wtkv16);
  pack(Wedge, 384, 128, 16, 32 * 256,  Wekv16);

  const int MT = NN / 16; // 625
  // proj: hA = relu(x @ Wp + bp)
  wmma_gemm_kernel<true, true, false><<<dim3(MT, HF / 16), 32, 0, stream>>>(
      x, 128, Wp16, 8, bp, nullptr, 0, hA, HF, 128);

  float* hin = hA;
  float* hout = hB;
  for (int L = 0; L < 2; ++L) {
    // xn = LN(hin) -> concat[:, :128]
    ln_kernel<<<NN, 128, 0, stream>>>(hin, g1w, b1n, concatB, 256);
    // kqv = xn @ Wkqv
    wmma_gemm_kernel<false, false, false><<<dim3(MT, KQV3 / 16), 32, 0, stream>>>(
        concatB, 256, Wkqv16, 24, nullptr, nullptr, 0, kqvB, KQV3, 128);
    // q = q_ + t_qq
    timeq_kernel<<<NN, 128, 0, stream>>>(times, Wt, bt, Wtime, kqvB, qbuf);
    // fused neighbor features + attention -> h2
    attn_kernel<<<NN, 256, 0, stream>>>(times, nbrs, rels, Wt, bt,
                                        Wtkv16, Wekv16, kqvB, qbuf,
                                        concatB, 256, h2B);
    // hn = LN(h2) -> concat[:, 128:]
    ln_kernel<<<NN, 128, 0, stream>>>(h2B, g2w, b2n, concatB + 128, 256);
    // z1 = relu(concat @ W1 + bl1)
    wmma_gemm_kernel<true, true, false><<<dim3(MT, HF / 16), 32, 0, stream>>>(
        concatB, 256, W116, 8, bl1, nullptr, 0, z1, HF, 256);
    // hout = z1 @ W2 + bl2 + h2
    wmma_gemm_kernel<true, false, true><<<dim3(MT, HF / 16), 32, 0, stream>>>(
        z1, HF, W216, 8, bl2, h2B, HF, hout, HF, 128);
    float* tmp = hin; hin = hout; hout = tmp;
  }

  // out = h @ Wo + bo
  wmma_gemm_kernel<true, false, false><<<dim3(MT, OUTF / 16), 32, 0, stream>>>(
      hin, HF, Wo16, 4, bo, nullptr, 0, out, OUTF, 128);
}
